// Attention_22325240004782
// MI455X (gfx1250) — matrix operand order repaired and checked by reading, not compiled, not run on hardware
//
#include <hip/hip_runtime.h>
#include <hip/hip_bf16.h>

// ---------- types ----------
typedef __attribute__((ext_vector_type(16))) __bf16        v16bf;
typedef __attribute__((ext_vector_type(8)))  float         v8f;
typedef __attribute__((ext_vector_type(4)))  float         v4f;
typedef __attribute__((ext_vector_type(4)))  unsigned int  v4u;
typedef __attribute__((ext_vector_type(2)))  unsigned int  v2u;

union FragBF {
    v16bf v;
    v4u   q[2];
};

// fp32 -> bf16 (round-to-nearest-even), two packed into a dword
__device__ __forceinline__ unsigned int pack2_bf16(float lo, float hi) {
    unsigned int a = __builtin_bit_cast(unsigned int, lo);
    unsigned int b = __builtin_bit_cast(unsigned int, hi);
    a = (a + 0x7FFFu + ((a >> 16) & 1u)) >> 16;
    b = (b + 0x7FFFu + ((b >> 16) & 1u)) >> 16;
    return (a & 0xFFFFu) | (b << 16);
}

#define Bsz 64
#define Lsz 196
#define Dsz 2048
#define Asz 1024
#define MROWS (Bsz * Lsz)        // 12544 = 196*64 = 392*32
#define KCHUNKS (Dsz / 32)       // 64

// ws layout: [ph: 64*1024 f32 = 256KB][Wsw: 2048*1024 bf16 swizzled = 4MB]
#define WS_PH_BYTES   (Bsz * Asz * 4)
#define WS_WSW_BYTES  ((size_t)Dsz * Asz * 2)
#define WS_NEED_FULL  (WS_PH_BYTES + WS_WSW_BYTES)

// =====================================================================
// Kernel 1: p_h[b,a] = h[b,:] @ W_h[:,a] + b_h[a] + b_att[a]
// =====================================================================
__global__ __launch_bounds__(256) void ph_kernel(
    const float* __restrict__ h, const float* __restrict__ Wh,
    const float* __restrict__ bh, const float* __restrict__ batt,
    float* __restrict__ ph) {
    int o = blockIdx.x * 256 + threadIdx.x;
    int b = o >> 10;
    int a = o & (Asz - 1);
    const float* hv = h + (size_t)b * Dsz;
    float s = 0.f;
#pragma unroll 4
    for (int k = 0; k < Dsz; ++k)
        s += hv[k] * Wh[(size_t)k * Asz + a];
    ph[o] = s + bh[a] + batt[a];
}

// =====================================================================
// Kernel 2: convert W_att fp32 -> bf16, pre-swizzled into the per-lane
// WMMA B-fragment layout so the GEMM loads it as contiguous b128 pairs.
// Layout: [kchunk 0..63][ntile 0..63][lane 0..31][pos 0..15] (bf16)
//   B frag (32x16, KxN), contiguous k order used here (same as A):  lane = (n&15) + 16*(kin>=16),  pos = kin&15
// =====================================================================
__global__ __launch_bounds__(256) void convw_kernel(
    const float* __restrict__ W, unsigned short* __restrict__ Wsw) {
    int t = blockIdx.x * 256 + threadIdx.x;  // 262144 tasks of 8 k-elements
    int n  = t & (Asz - 1);
    int kr = t >> 10;                         // 8-k run id, 0..255
    const float* src = W + (size_t)(kr * 8) * Asz + n;   // stride Asz over k
    v4u p;
    p.x = pack2_bf16(src[0 * Asz], src[1 * Asz]);
    p.y = pack2_bf16(src[2 * Asz], src[3 * Asz]);
    p.z = pack2_bf16(src[4 * Asz], src[5 * Asz]);
    p.w = pack2_bf16(src[6 * Asz], src[7 * Asz]);
    int kc    = kr >> 2;
    int kin0  = (kr & 3) * 8;                 // 0,8,16,24
    int lane  = (n & 15) + (kin0 & 16);
    int pos0  = kin0 & 15;                    // 0 or 8
    size_t off = (((size_t)kc * 64 + (n >> 4)) * 32 + lane) * 16 + pos0;
    *(v4u*)&Wsw[off] = p;
}

// =====================================================================
// Kernel 3 (main): fused scores GEMM.
// Block = 64 rows x all 1024 cols, 512 threads = 16 waves.
// Wave w: rows 0..63 (4 mtiles) x cols [w*64, w*64+64) (4 ntiles)
//   -> 16 acc tiles (128 VGPRs), 16 v_wmma per K-step, 64 K-steps.
// B frags: direct from pre-swizzled global bf16 (L2 resident, no LDS).
// A tiles: fp32 af -> bf16, staged in double-buffered LDS (1 barrier/step).
// =====================================================================
__device__ __forceinline__ void stageA(unsigned short* __restrict__ dst,
                                       const float* __restrict__ af,
                                       int row0, int k0, int tid) {
    // A frag k order used here (contiguous, same as B): lane = (m&15) + 16*((k>>4)&1),
    //                                 pos  = k&15
    int g = tid & 7;          // 4-k group
    int m = tid >> 3;         // row in tile 0..63
    v4f f = *(const v4f*)(af + (size_t)(row0 + m) * Dsz + k0 + g * 4);
    v2u p;
    p.x = pack2_bf16(f.x, f.y);
    p.y = pack2_bf16(f.z, f.w);
    int mt   = m >> 4;
    int lane = (m & 15) + (((unsigned)g >> 2) << 4);
    int pos  = (g & 3) << 2;
    *(v2u*)&dst[mt * 512 + lane * 16 + pos] = p;
}

__global__ __launch_bounds__(512) void scores_wmma_kernel(
    const float* __restrict__ af,            // [12544, 2048] fp32
    const unsigned short* __restrict__ Wsw,  // swizzled bf16 W_att
    const float* __restrict__ ph,            // [64, 1024] (incl. b_h + b_att)
    const float* __restrict__ walpha,        // [1024]
    const float* __restrict__ balpha,        // [1]
    float* __restrict__ scores) {            // [12544]
    __shared__ __align__(16) unsigned short lds_a[2][4 * 512];  // 8 KB dbl-buf
    __shared__ float lds_part[16][64];

    const int tid  = threadIdx.x;
    const int lane = tid & 31;
    const int wave = tid >> 5;          // 0..15
    const int row0 = blockIdx.x * 64;

    v8f acc[4][4];
#pragma unroll
    for (int i = 0; i < 4; ++i)
#pragma unroll
        for (int j = 0; j < 4; ++j)
#pragma unroll
            for (int r = 0; r < 8; ++r) acc[i][j][r] = 0.f;

    float wa[4];
#pragma unroll
    for (int i = 0; i < 4; ++i)
        wa[i] = walpha[(wave * 4 + i) * 16 + (lane & 15)];

    stageA(lds_a[0], af, row0, 0, tid);
    __syncthreads();

    for (int kc = 0; kc < KCHUNKS; ++kc) {
        if (kc + 1 < KCHUNKS)
            stageA(lds_a[(kc + 1) & 1], af, row0, (kc + 1) * 32, tid);

        // B fragments: contiguous 32B per lane from pre-swizzled W (L2)
        const unsigned short* bp =
            Wsw + (((size_t)kc * 64 + wave * 4) * 32 + lane) * 16;
        if (kc + 1 < KCHUNKS) {
#pragma unroll
            for (int i = 0; i < 4; ++i)
                __builtin_prefetch((const void*)(bp + 64 * 512 + i * 512), 0, 1);
        }
        FragBF bb[4];
#pragma unroll
        for (int i = 0; i < 4; ++i) {
            bb[i].q[0] = *(const v4u*)(bp + i * 512);
            bb[i].q[1] = *(const v4u*)(bp + i * 512 + 8);
        }
        const unsigned short* ap = &lds_a[kc & 1][lane * 16];
        FragBF aa[4];
#pragma unroll
        for (int mt = 0; mt < 4; ++mt) {
            aa[mt].q[0] = *(const v4u*)(ap + mt * 512);
            aa[mt].q[1] = *(const v4u*)(ap + mt * 512 + 8);
        }
#pragma unroll
        for (int mt = 0; mt < 4; ++mt)
#pragma unroll
            for (int i = 0; i < 4; ++i)
                acc[mt][i] = __builtin_amdgcn_wmma_f32_16x16x32_bf16(
                    false, aa[mt].v, false, bb[i].v, (short)0, acc[mt][i],
                    false, false);
        __syncthreads();
    }

    // ---- epilogue: +ph, tanh, *w_alpha, reduce over cols ----
    const float bal = balpha[0];
#pragma unroll
    for (int mt = 0; mt < 4; ++mt) {
#pragma unroll
        for (int r = 0; r < 8; ++r) {
            int rib = mt * 16 + r + ((lane >> 4) << 3);
            int row = row0 + rib;
            int bidx = row / Lsz;
            const float* phb = ph + (size_t)bidx * Asz;
            float s = 0.f;
#pragma unroll
            for (int i = 0; i < 4; ++i) {
                int col = (wave * 4 + i) * 16 + (lane & 15);
                float v = acc[mt][i][r] + phb[col];
                s += tanhf(v) * wa[i];
            }
#pragma unroll
            for (int d = 1; d < 16; d <<= 1) s += __shfl_xor(s, d, 32);
            if ((lane & 15) == 0) lds_part[wave][rib] = s;
        }
    }
    __syncthreads();
    if (tid < 64) {
        float s = bal;
#pragma unroll
        for (int w = 0; w < 16; ++w) s += lds_part[w][tid];
        scores[row0 + tid] = s;
    }
}

// =====================================================================
// Fallback scores kernel (no extra ws needed beyond ph): 32-row blocks,
// B staged+converted through LDS.
// =====================================================================
__global__ __launch_bounds__(256) void scores_fallback_kernel(
    const float* __restrict__ af, const float* __restrict__ Watt,
    const float* __restrict__ ph, const float* __restrict__ walpha,
    const float* __restrict__ balpha, float* __restrict__ scores) {
    __shared__ __align__(16) unsigned short lds_a[2 * 512];
    __shared__ __align__(16) unsigned short lds_b[32 * 512];
    __shared__ float lds_part[8][32];

    const int tid  = threadIdx.x;
    const int lane = tid & 31;
    const int wave = tid >> 5;
    const int row0 = blockIdx.x * 32;

    v8f acc[2][8];
#pragma unroll
    for (int i = 0; i < 2; ++i)
#pragma unroll
        for (int j = 0; j < 8; ++j)
#pragma unroll
            for (int r = 0; r < 8; ++r) acc[i][j][r] = 0.f;

    float wa[2][4];
#pragma unroll
    for (int hf = 0; hf < 2; ++hf)
#pragma unroll
        for (int i = 0; i < 4; ++i)
            wa[hf][i] = walpha[hf * 512 + (wave * 4 + i) * 16 + (lane & 15)];

    for (int k0 = 0; k0 < Dsz; k0 += 32) {
        __syncthreads();
        if (tid < 128) {
            int g = tid & 3;
            int m = tid >> 2;
            const float* src = af + (size_t)(row0 + m) * Dsz + k0 + g * 8;
            v4f f0 = *(const v4f*)src;
            v4f f1 = *(const v4f*)(src + 4);
            v4u p;
            p.x = pack2_bf16(f0.x, f0.y); p.y = pack2_bf16(f0.z, f0.w);
            p.z = pack2_bf16(f1.x, f1.y); p.w = pack2_bf16(f1.z, f1.w);
            int mt  = m >> 4;
            int ln  = (m & 15) + ((g >> 1) << 4);
            int pos = (g & 1) << 3;
            *(v4u*)&lds_a[mt * 512 + ln * 16 + pos] = p;
        }
        for (int hf = 0; hf < 2; ++hf) {
            if (hf == 1) __syncthreads();
#pragma unroll
            for (int it = 0; it < 8; ++it) {
                int task = it * 256 + tid;
                int g  = task >> 9;
                int nl = task & 511;
                const float* src = Watt + (size_t)(k0 + g * 8) * Asz + hf * 512 + nl;
                v4u p;
                p.x = pack2_bf16(src[0 * Asz], src[1 * Asz]);
                p.y = pack2_bf16(src[2 * Asz], src[3 * Asz]);
                p.z = pack2_bf16(src[4 * Asz], src[5 * Asz]);
                p.w = pack2_bf16(src[6 * Asz], src[7 * Asz]);
                int ln  = (nl & 15) + ((g & 2) << 3);
                int pos = (g & 1) << 3;
                *(v4u*)&lds_b[(nl >> 4) * 512 + ln * 16 + pos] = p;
            }
            __syncthreads();
            FragBF a0, a1;
            a0.q[0] = *(const v4u*)&lds_a[0 * 512 + lane * 16];
            a0.q[1] = *(const v4u*)&lds_a[0 * 512 + lane * 16 + 8];
            a1.q[0] = *(const v4u*)&lds_a[1 * 512 + lane * 16];
            a1.q[1] = *(const v4u*)&lds_a[1 * 512 + lane * 16 + 8];
#pragma unroll
            for (int i = 0; i < 4; ++i) {
                int ntl = wave * 4 + i;
                FragBF bb;
                bb.q[0] = *(const v4u*)&lds_b[ntl * 512 + lane * 16];
                bb.q[1] = *(const v4u*)&lds_b[ntl * 512 + lane * 16 + 8];
                acc[0][hf * 4 + i] = __builtin_amdgcn_wmma_f32_16x16x32_bf16(
                    false, a0.v, false, bb.v, (short)0, acc[0][hf * 4 + i], false, false);
                acc[1][hf * 4 + i] = __builtin_amdgcn_wmma_f32_16x16x32_bf16(
                    false, a1.v, false, bb.v, (short)0, acc[1][hf * 4 + i], false, false);
            }
        }
    }
    __syncthreads();
    const float bal = balpha[0];
#pragma unroll
    for (int mt = 0; mt < 2; ++mt) {
#pragma unroll
        for (int r = 0; r < 8; ++r) {
            int rib = mt * 16 + r + ((lane >> 4) << 3);
            int row = row0 + rib;
            int bidx = row / Lsz;
            const float* phb = ph + (size_t)bidx * Asz;
            float s = 0.f;
#pragma unroll
            for (int hf = 0; hf < 2; ++hf)
#pragma unroll
                for (int i = 0; i < 4; ++i) {
                    int col = hf * 512 + (wave * 4 + i) * 16 + (lane & 15);
                    float v = acc[mt][hf * 4 + i][r] + phb[col];
                    s += tanhf(v) * wa[hf][i];
                }
#pragma unroll
            for (int d = 1; d < 16; d <<= 1) s += __shfl_xor(s, d, 32);
            if ((lane & 15) == 0) lds_part[wave][rib] = s;
        }
    }
    __syncthreads();
    if (tid < 32) {
        float s = bal;
#pragma unroll
        for (int w = 0; w < 8; ++w) s += lds_part[w][tid];
        scores[row0 + tid] = s;
    }
}

// =====================================================================
// Kernel 4: in-place softmax over L=196 per batch
// =====================================================================
__global__ __launch_bounds__(256) void softmax_kernel(float* __restrict__ sw) {
    __shared__ float red[8];
    int b = blockIdx.x, t = threadIdx.x;
    int lane = t & 31, wv = t >> 5;
    float v = (t < Lsz) ? sw[b * Lsz + t] : -3.0e38f;
    float m = v;
#pragma unroll
    for (int d = 16; d; d >>= 1) m = fmaxf(m, __shfl_xor(m, d, 32));
    if (lane == 0) red[wv] = m;
    __syncthreads();
    m = red[0];
#pragma unroll
    for (int i = 1; i < 8; ++i) m = fmaxf(m, red[i]);
    float e = (t < Lsz) ? __expf(v - m) : 0.f;
    float s = e;
#pragma unroll
    for (int d = 16; d; d >>= 1) s += __shfl_xor(s, d, 32);
    __syncthreads();
    if (lane == 0) red[wv] = s;
    __syncthreads();
    s = 0.f;
#pragma unroll
    for (int i = 0; i < 8; ++i) s += red[i];
    if (t < Lsz) sw[b * Lsz + t] = e / s;
}

// =====================================================================
// Kernel 5: weighted[b,d] = sum_l w[b,l] * af[b,l,d]   (bandwidth bound)
// =====================================================================
__global__ __launch_bounds__(256) void weighted_kernel(
    const float* __restrict__ af, const float* __restrict__ weights,
    float* __restrict__ out) {
    __shared__ float w[Lsz];
    int b = blockIdx.y;
    int d = blockIdx.x * 256 + threadIdx.x;
    if (threadIdx.x < Lsz) w[threadIdx.x] = weights[b * Lsz + threadIdx.x];
    __syncthreads();
    const float* p = af + (size_t)b * Lsz * Dsz + d;
    float s = 0.f;
#pragma unroll 4
    for (int l = 0; l < Lsz; ++l) s += w[l] * p[(size_t)l * Dsz];
    out[(size_t)b * Dsz + d] = s;
}

// =====================================================================
extern "C" void kernel_launch(void* const* d_in, const int* in_sizes, int n_in,
                              void* d_out, int out_size, void* d_ws, size_t ws_size,
                              hipStream_t stream) {
    const float* af    = (const float*)d_in[0];
    const float* h     = (const float*)d_in[1];
    const float* W_att = (const float*)d_in[2];
    const float* b_att = (const float*)d_in[3];
    const float* W_h   = (const float*)d_in[4];
    const float* b_h   = (const float*)d_in[5];
    const float* w_al  = (const float*)d_in[6];
    const float* b_al  = (const float*)d_in[7];

    float* out      = (float*)d_out;
    float* weighted = out;
    float* weights  = out + (size_t)Bsz * Dsz;     // scores live here first
    float* ph       = (float*)d_ws;
    unsigned short* Wsw = (unsigned short*)((char*)d_ws + WS_PH_BYTES);

    ph_kernel<<<(Bsz * Asz) / 256, 256, 0, stream>>>(h, W_h, b_h, b_att, ph);

    if (ws_size >= WS_NEED_FULL) {
        convw_kernel<<<(Dsz * Asz / 8) / 256, 256, 0, stream>>>(W_att, Wsw);
        scores_wmma_kernel<<<MROWS / 64, 512, 0, stream>>>(
            af, Wsw, ph, w_al, b_al, weights);
    } else {
        scores_fallback_kernel<<<MROWS / 32, 256, 0, stream>>>(
            af, W_att, ph, w_al, b_al, weights);
    }

    softmax_kernel<<<Bsz, 256, 0, stream>>>(weights);
    weighted_kernel<<<dim3(Dsz / 256, Bsz), 256, 0, stream>>>(af, weights, weighted);
}
